// OADModel_91242285236740
// MI455X (gfx1250) — matrix operand order repaired and checked by reading, not compiled, not run on hardware
//
#include <hip/hip_runtime.h>

typedef _Float16 f16;
typedef __attribute__((ext_vector_type(16))) _Float16 v16h;
typedef __attribute__((ext_vector_type(8)))  _Float16 v8h;
typedef __attribute__((ext_vector_type(8)))  float    v8f;
typedef __attribute__((ext_vector_type(4)))  unsigned u32x4;
typedef __attribute__((ext_vector_type(8)))  unsigned u32x8;

union V16H { v16h v; v8h h[2]; };

// ---------------------------------------------------------------- helpers ---

__device__ inline float gelu_erf(float x) {
    return 0.5f * x * (1.0f + erff(x * 0.70710678118654752440f));
}

__device__ inline float sigmoidf(float x) { return 1.0f / (1.0f + expf(-x)); }

__device__ inline v8f wmma_f16(v16h a, v16h b, v8f c) {
    return __builtin_amdgcn_wmma_f32_16x16x32_f16(
        /*neg_a=*/false, a, /*neg_b=*/false, b,
        /*c_mod=*/(short)0, c, /*reuse_a=*/false, /*reuse_b=*/false);
}

// A fragment: A is [16 x lda] f16 row-major. lane m = lane&15 holds row m.
// elements 0..7 -> K = k0 + hi*8 + e ; elements 8..15 -> K = k0 + hi*8 + 16 + e
__device__ inline v16h ldA(const f16* A, int lda, int k0, int lane) {
    int m = lane & 15, hi = lane >> 4;
    const f16* p = A + (size_t)m * lda + k0 + hi * 8;
    V16H r;
    r.h[0] = *(const v8h*)(p);
    r.h[1] = *(const v8h*)(p + 16);
    return r.v;
}

// B fragment from B^T stored [N x K] row-major. n = caller-provided column.
// elements 0..7 -> K = k0 + hi*8 + e ; elements 8..15 -> K = k0 + hi*8 + 16 + e
__device__ inline v16h ldB(const f16* BT, int K, int n, int k0, int hi) {
    const f16* p = BT + (size_t)n * K + k0 + hi * 8;
    V16H r;
    r.h[0] = *(const v8h*)(p);
    r.h[1] = *(const v8h*)(p + 16);
    return r.v;
}

__device__ inline float wave_sum(float v) {
#pragma unroll
    for (int m = 16; m >= 1; m >>= 1) v += __shfl_xor(v, m, 32);
    return v;
}

// On gfx1250 the low 32 bits of a generic pointer into LDS are the LDS byte
// address.
__device__ inline unsigned lds_addr32(const void* p) {
    return (unsigned)(size_t)p;
}

// ------------------------------------------------------------ prep kernels ---

__global__ void cast_kernel(const float* __restrict__ s, f16* __restrict__ d, long n) {
    long i = (long)blockIdx.x * blockDim.x + threadIdx.x;
    if (i < n) d[i] = (f16)s[i];
}

// src [R][C] fp32 -> dst [C][R] f16   (produces N x K "BT" layouts)
__global__ void transpose_cast_kernel(const float* __restrict__ s, f16* __restrict__ d,
                                      int R, int C) {
    long i = (long)blockIdx.x * blockDim.x + threadIdx.x;
    if (i >= (long)R * C) return;
    int r = (int)(i / C), c = (int)(i % C);
    d[(size_t)c * R + r] = (f16)s[i];
}

__global__ void zero_kernel(float* p, int n) {
    int i = blockIdx.x * blockDim.x + threadIdx.x;
    if (i < n) p[i] = 0.0f;
}

// -------------------------------------------------- pre-layer (all T fused) ---
// preact[t][b][e] = gelu(LN(feat[b][t][:] @ pre_W + pre_b))
// grid: 512 blocks (one 16-row M-tile of the 8192-row GEMM), 256 threads (8 waves).
// The 16x2048 fp32 A-tile is DMA'd into LDS by the Tensor Data Mover.
__global__ void pre_kernel(const float* __restrict__ feat, const f16* __restrict__ preWT,
                           const float* __restrict__ pb, const float* __restrict__ g,
                           const float* __restrict__ bb, float* __restrict__ preact) {
    __shared__ __align__(16) float atile[16][2048];   // 128 KB (TDM destination)
    __shared__ __align__(16) float y[16][1024];       // 64 KB
    __shared__ float mean[16], rstd[16];
    int tid = threadIdx.x, lane = tid & 31, wave = tid >> 5;
    int m = lane & 15, hi = lane >> 4, ln_ = lane & 15;
    int ncol0 = wave * 128;

    // --- TDM: tile load feat[blk*16 .. blk*16+15][0..2047] (fp32) into LDS ---
    if (wave == 0) {
        unsigned long long ga = (unsigned long long)(feat + (size_t)blockIdx.x * 16 * 2048);
        u32x4 g0;
        g0[0] = 1u;                                       // count=1, user mode
        g0[1] = lds_addr32(&atile[0][0]);                 // lds_addr
        g0[2] = (unsigned)(ga & 0xffffffffu);             // global_addr[31:0]
        g0[3] = (unsigned)((ga >> 32) & 0x01ffffffu) | (2u << 30);  // addr[56:32] | type=2
        u32x8 g1;
        g1[0] = (2u << 16);        // workgroup_mask=0, data_size=2 (4 bytes)
        g1[1] = (2048u << 16);     // tensor_dim0[15:0] in bits 63:48
        g1[2] = (16u << 16);       // tensor_dim0[31:16]=0 | tensor_dim1[15:0]=16
        g1[3] = (2048u << 16);     // tensor_dim1[31:16]=0 | tile_dim0=2048
        g1[4] = 16u;               // tile_dim1=16, tile_dim2=0
        g1[5] = 2048u;             // tensor_dim0_stride[31:0]
        g1[6] = 0u;                // stride hi | tensor_dim1_stride lo
        g1[7] = 0u;
        asm volatile("tensor_load_to_lds %0, %1" :: "s"(g0), "s"(g1) : "memory");
        __builtin_amdgcn_s_wait_tensorcnt(0);
    }
    __syncthreads();

    v8f z = {0,0,0,0,0,0,0,0};
    v8f acc[8];
#pragma unroll
    for (int t = 0; t < 8; ++t) acc[t] = z;

    const float* arow = &atile[m][0];
    for (int k0 = 0; k0 < 2048; k0 += 32) {
        V16H a;
        const float* p = arow + k0 + hi * 8;
#pragma unroll
        for (int e = 0; e < 8; ++e) { a.h[0][e] = (f16)p[e]; a.h[1][e] = (f16)p[e + 16]; }
#pragma unroll
        for (int t = 0; t < 8; ++t) {
            v16h w = ldB(preWT, 2048, ncol0 + t * 16 + ln_, k0, hi);
            acc[t] = wmma_f16(a.v, w, acc[t]);
        }
    }
#pragma unroll
    for (int t = 0; t < 8; ++t) {
        int col = ncol0 + t * 16 + ln_;
#pragma unroll
        for (int r = 0; r < 8; ++r) y[r + hi * 8][col] = acc[t][r] + pb[col];
    }
    __syncthreads();
#pragma unroll
    for (int rr = 0; rr < 2; ++rr) {
        int r = wave * 2 + rr;
        float s = 0.f, s2 = 0.f;
        for (int e = lane; e < 1024; e += 32) { float v = y[r][e]; s += v; s2 += v * v; }
        s = wave_sum(s); s2 = wave_sum(s2);
        if (lane == 0) {
            float mu = s * (1.0f / 1024.0f);
            mean[r] = mu;
            rstd[r] = rsqrtf(s2 * (1.0f / 1024.0f) - mu * mu + 1e-5f);
        }
    }
    __syncthreads();
    for (int idx = tid; idx < 16 * 1024; idx += 256) {
        int r = idx >> 10, e = idx & 1023;
        size_t gr = (size_t)blockIdx.x * 16 + r;
        int b = (int)(gr >> 9), t = (int)(gr & 511);
        float v = (y[r][e] - mean[r]) * rstd[r] * g[e] + bb[e];
        preact[((size_t)t * 16 + b) * 1024 + e] = gelu_erf(v);
    }
}

// ------------------------------------------------------- fused GRU layer -----
// One launch per (t, layer). grid 16 blocks x 128 threads (4 waves).
// Block handles 64 output cols; wave handles 16 cols -> 6 WMMA tiles (gx/gh x r/z/n).
// Software-pipelined: next k-step's fragments load while current WMMAs issue.
__global__ void gru_kernel(const float* __restrict__ x, const float* __restrict__ hprev,
                           float* __restrict__ hnew, f16* __restrict__ h16,
                           const f16* __restrict__ WihT, const f16* __restrict__ WhhT,
                           const float* __restrict__ bih, const float* __restrict__ bhh,
                           const float* __restrict__ xg, const float* __restrict__ xb,
                           const float* __restrict__ hg, const float* __restrict__ hb) {
    __shared__ __align__(16) f16 nx[16][1024];
    __shared__ __align__(16) f16 nh[16][1024];
    int tid = threadIdx.x, lane = tid & 31, wave = tid >> 5;
    int ln_ = lane & 15, hi = lane >> 4;

    // LayerNorm x and h into f16 LDS (4 rows per wave, wave32 butterfly)
#pragma unroll
    for (int rr = 0; rr < 4; ++rr) {
        int r = wave * 4 + rr;
        const float* xr = x + (size_t)r * 1024;
        const float* hr = hprev + (size_t)r * 1024;
        float sx = 0.f, sx2 = 0.f, sh = 0.f, sh2 = 0.f;
        for (int e = lane; e < 1024; e += 32) {
            float v = xr[e]; sx += v; sx2 += v * v;
            float w = hr[e]; sh += w; sh2 += w * w;
        }
        sx = wave_sum(sx); sx2 = wave_sum(sx2);
        sh = wave_sum(sh); sh2 = wave_sum(sh2);
        float mx = sx * (1.0f / 1024.0f);
        float rx = rsqrtf(sx2 * (1.0f / 1024.0f) - mx * mx + 1e-5f);
        float mh = sh * (1.0f / 1024.0f);
        float rh = rsqrtf(sh2 * (1.0f / 1024.0f) - mh * mh + 1e-5f);
        for (int e = lane; e < 1024; e += 32) {
            nx[r][e] = (f16)(((xr[e] - mx) * rx) * xg[e] + xb[e]);
            nh[r][e] = (f16)(((hr[e] - mh) * rh) * hg[e] + hb[e]);
        }
    }
    __syncthreads();

    int jt = blockIdx.x * 64 + wave * 16;       // column tile base in [0,1024)
    int jr0 = jt + ln_, jz0 = 1024 + jt + ln_, jn0 = 2048 + jt + ln_;
    const f16* nxp = &nx[0][0];
    const f16* nhp = &nh[0][0];

    v8f z = {0,0,0,0,0,0,0,0};
    v8f axr = z, axz = z, axn = z, ahr = z, ahz = z, ahn = z;

    v16h ax  = ldA(nxp, 1024, 0, lane);
    v16h ah  = ldA(nhp, 1024, 0, lane);
    v16h bxr = ldB(WihT, 1024, jr0, 0, hi);
    v16h bxz = ldB(WihT, 1024, jz0, 0, hi);
    v16h bxn = ldB(WihT, 1024, jn0, 0, hi);
    v16h bhr = ldB(WhhT, 1024, jr0, 0, hi);
    v16h bhz = ldB(WhhT, 1024, jz0, 0, hi);
    v16h bhn = ldB(WhhT, 1024, jn0, 0, hi);

    for (int k0 = 0; k0 < 1024; k0 += 32) {
        int kn = (k0 + 32) & 1023;              // prefetch wraps to 0 on last iter
        v16h ax2 = ldA(nxp, 1024, kn, lane);
        v16h ah2 = ldA(nhp, 1024, kn, lane);
        v16h t0 = ldB(WihT, 1024, jr0, kn, hi);
        v16h t1 = ldB(WihT, 1024, jz0, kn, hi);
        v16h t2 = ldB(WihT, 1024, jn0, kn, hi);
        v16h t3 = ldB(WhhT, 1024, jr0, kn, hi);
        v16h t4 = ldB(WhhT, 1024, jz0, kn, hi);
        v16h t5 = ldB(WhhT, 1024, jn0, kn, hi);
        axr = wmma_f16(ax, bxr, axr);
        axz = wmma_f16(ax, bxz, axz);
        axn = wmma_f16(ax, bxn, axn);
        ahr = wmma_f16(ah, bhr, ahr);
        ahz = wmma_f16(ah, bhz, ahz);
        ahn = wmma_f16(ah, bhn, ahn);
        ax = ax2; ah = ah2;
        bxr = t0; bxz = t1; bxn = t2; bhr = t3; bhz = t4; bhn = t5;
    }

    int col = jt + ln_;
    float br = bih[col],        bhr_ = bhh[col];
    float bz = bih[1024 + col], bhz_ = bhh[1024 + col];
    float bn = bih[2048 + col], bhn_ = bhh[2048 + col];
#pragma unroll
    for (int r = 0; r < 8; ++r) {
        int m = r + hi * 8;
        float rg = sigmoidf((axr[r] + br) + (ahr[r] + bhr_));
        float zg = sigmoidf((axz[r] + bz) + (ahz[r] + bhz_));
        float ng = tanhf((axn[r] + bn) + rg * (ahn[r] + bhn_));
        float nhv = (float)nh[m][col];
        float hv = (1.0f - zg) * ng + zg * nhv;
        hnew[(size_t)m * 1024 + col] = hv;
        h16[(size_t)m * 1024 + col] = (f16)hv;
    }
}

// -------------------------------------------------- proj: tokens = h @ projW ---
// grid (8, 4=layer) x 128 threads. tokens [16][4][512] fp32.
__global__ void proj_kernel(const f16* __restrict__ h16, const f16* __restrict__ projT,
                            const float* __restrict__ pb, float* __restrict__ tokens) {
    int l = blockIdx.y;
    int tid = threadIdx.x, lane = tid & 31, wave = tid >> 5;
    int ln_ = lane & 15, hi = lane >> 4;
    int n = blockIdx.x * 64 + wave * 16 + ln_;
    const f16* A = h16 + (size_t)l * 16 * 1024;
    v8f acc = {0,0,0,0,0,0,0,0};
    for (int k0 = 0; k0 < 1024; k0 += 32)
        acc = wmma_f16(ldA(A, 1024, k0, lane), ldB(projT, 1024, n, k0, hi), acc);
#pragma unroll
    for (int r = 0; r < 8; ++r) {
        int m = r + hi * 8;
        tokens[(size_t)m * 2048 + l * 512 + n] = acc[r] + pb[n];
    }
}

// ----------------------------------------------- token mixer (tiny, scalar) ---
// grid 16 (=b) x 256. Token block staged into LDS with async-to-LDS DMA.
__global__ void tokmix_kernel(float* __restrict__ tokens,
                              const float* __restrict__ g1, const float* __restrict__ b1,
                              const float* __restrict__ W1, const float* __restrict__ tb1,
                              const float* __restrict__ W2, const float* __restrict__ tb2) {
    __shared__ __align__(16) float tk[4][512];
    __shared__ float mean[4], rstd[4];
    int b = blockIdx.x, tid = threadIdx.x, lane = tid & 31, wave = tid >> 5;
    float* base = tokens + (size_t)b * 2048;

    // async stage: 512 floats (tk[0] only) = 128 x b128, issued by waves 0-3
    if (tid < 128) {
        unsigned lds = lds_addr32(&tk[0][0]) + (unsigned)tid * 16u;
        unsigned long long ga = (unsigned long long)(base + tid * 4);
        asm volatile("global_load_async_to_lds_b128 %0, %1, off"
                     :: "v"(lds), "v"(ga) : "memory");
    }
    asm volatile("s_wait_asynccnt 0x0" ::: "memory");
    __syncthreads();

    if (wave < 4) {
        int l = wave;
        float s = 0.f, s2 = 0.f;
        for (int e = lane; e < 512; e += 32) { float v = tk[l][e]; s += v; s2 += v * v; }
        s = wave_sum(s); s2 = wave_sum(s2);
        if (lane == 0) {
            float mu = s * (1.0f / 512.0f);
            mean[l] = mu;
            rstd[l] = rsqrtf(s2 * (1.0f / 512.0f) - mu * mu + 1e-5f);
        }
    }
    __syncthreads();
    for (int p = tid; p < 512; p += 256) {
        float tl[4], add[4];
#pragma unroll
        for (int l = 0; l < 4; ++l) {
            tl[l] = (tk[l][p] - mean[l]) * rstd[l] * g1[p] + b1[p];
            add[l] = tb2[l];
        }
#pragma unroll
        for (int k = 0; k < 16; ++k) {
            float uk = tb1[k];
#pragma unroll
            for (int l = 0; l < 4; ++l) uk += tl[l] * W1[l * 16 + k];
            uk = gelu_erf(uk);
#pragma unroll
            for (int l = 0; l < 4; ++l) add[l] += uk * W2[k * 4 + l];
        }
#pragma unroll
        for (int l = 0; l < 4; ++l) base[l * 512 + p] = tk[l][p] + add[l];
    }
}

// ----------------------------------------- per-row LN -> f16 (channel mixer) ---
// grid 64 (row = b*4+l) x 32 (one wave).
__global__ void lnrow_kernel(const float* __restrict__ X, const float* __restrict__ g,
                             const float* __restrict__ b, f16* __restrict__ out) {
    int row = blockIdx.x, lane = threadIdx.x;
    const float* xr = X + (size_t)row * 512;
    float s = 0.f, s2 = 0.f;
    for (int e = lane; e < 512; e += 32) { float v = xr[e]; s += v; s2 += v * v; }
    s = wave_sum(s); s2 = wave_sum(s2);
    float mu = s * (1.0f / 512.0f);
    float r = rsqrtf(s2 * (1.0f / 512.0f) - mu * mu + 1e-5f);
    for (int e = lane; e < 512; e += 32)
        out[(size_t)row * 512 + e] = (f16)((xr[e] - mu) * r * g[e] + b[e]);
}

// ----------------------------------------------------- channel MLP GEMM 1 ----
// u16[64][2048] = f16(gelu(t16 @ chW1 + b1)). grid (32, 4=Mtile) x 128.
__global__ void mlp1_kernel(const f16* __restrict__ t16, const f16* __restrict__ W1T,
                            const float* __restrict__ b1, f16* __restrict__ u16) {
    int tid = threadIdx.x, lane = tid & 31, wave = tid >> 5;
    int ln_ = lane & 15, hi = lane >> 4;
    int n = blockIdx.x * 64 + wave * 16 + ln_;
    int mt = blockIdx.y;
    const f16* A = t16 + (size_t)mt * 16 * 512;
    v8f acc = {0,0,0,0,0,0,0,0};
    for (int k0 = 0; k0 < 512; k0 += 32)
        acc = wmma_f16(ldA(A, 512, k0, lane), ldB(W1T, 512, n, k0, hi), acc);
#pragma unroll
    for (int r = 0; r < 8; ++r) {
        int m = mt * 16 + r + hi * 8;
        u16[(size_t)m * 2048 + n] = (f16)gelu_erf(acc[r] + b1[n]);
    }
}

// ----------------------------------------------------- channel MLP GEMM 2 ----
// tokens[64][512] += u16 @ chW2 + b2. grid (8, 4=Mtile) x 128.
__global__ void mlp2_kernel(const f16* __restrict__ u16, const f16* __restrict__ W2T,
                            const float* __restrict__ b2, float* __restrict__ tokens) {
    int tid = threadIdx.x, lane = tid & 31, wave = tid >> 5;
    int ln_ = lane & 15, hi = lane >> 4;
    int n = blockIdx.x * 64 + wave * 16 + ln_;
    int mt = blockIdx.y;
    const f16* A = u16 + (size_t)mt * 16 * 2048;
    v8f acc = {0,0,0,0,0,0,0,0};
    for (int k0 = 0; k0 < 2048; k0 += 32)
        acc = wmma_f16(ldA(A, 2048, k0, lane), ldB(W2T, 2048, n, k0, hi), acc);
#pragma unroll
    for (int r = 0; r < 8; ++r) {
        int m = mt * 16 + r + hi * 8;
        tokens[(size_t)m * 512 + n] += acc[r] + b2[n];
    }
}

// ------------------------------------------- final LN + mean-pool + head -----
// grid 16 (=b) x 256. logits[b][t][c].
__global__ void head_kernel(const float* __restrict__ tokens, const float* __restrict__ fg,
                            const float* __restrict__ fb, const float* __restrict__ hW,
                            const float* __restrict__ hb, float* __restrict__ out, int t) {
    __shared__ float pooled[512];
    __shared__ float mean[4], rstd[4];
    int b = blockIdx.x, tid = threadIdx.x, lane = tid & 31, wave = tid >> 5;
    const float* tb_ = tokens + (size_t)b * 2048;
    if (wave < 4) {
        int l = wave;
        float s = 0.f, s2 = 0.f;
        for (int e = lane; e < 512; e += 32) { float v = tb_[l * 512 + e]; s += v; s2 += v * v; }
        s = wave_sum(s); s2 = wave_sum(s2);
        if (lane == 0) {
            float mu = s * (1.0f / 512.0f);
            mean[l] = mu;
            rstd[l] = rsqrtf(s2 * (1.0f / 512.0f) - mu * mu + 1e-5f);
        }
    }
    __syncthreads();
    for (int p = tid; p < 512; p += 256) {
        float s = 0.f;
#pragma unroll
        for (int l = 0; l < 4; ++l) s += (tb_[l * 512 + p] - mean[l]) * rstd[l];
        pooled[p] = fg[p] * s * 0.25f + fb[p];
    }
    __syncthreads();
    if (tid < 22) {
        float acc = hb[tid];
        for (int p = 0; p < 512; ++p) acc += pooled[p] * hW[p * 22 + tid];
        out[((size_t)b * 512 + t) * 22 + tid] = acc;
    }
}

// ------------------------------------------------------- weighted CE loss ----
__global__ void loss_kernel(const float* __restrict__ logits, const int* __restrict__ t1,
                            const int* __restrict__ t2, const float* __restrict__ w,
                            float* __restrict__ loss_out) {
    __shared__ float r0[256], r1[256], r2[256], r3[256];
    int tid = threadIdx.x;
    float n1 = 0.f, d1 = 0.f, n2 = 0.f, d2 = 0.f;
    for (int row = tid; row < 16 * 512; row += 256) {
        const float* lr = logits + (size_t)row * 22;
        float mx = lr[0];
        for (int c = 1; c < 22; ++c) mx = fmaxf(mx, lr[c]);
        float se = 0.f;
        for (int c = 0; c < 22; ++c) se += expf(lr[c] - mx);
        float lse = mx + logf(se);
        int a = t1[row]; float wa = w[a]; n1 += wa * (lse - lr[a]); d1 += wa;
        int b = t2[row]; float wb = w[b]; n2 += wb * (lse - lr[b]); d2 += wb;
    }
    r0[tid] = n1; r1[tid] = d1; r2[tid] = n2; r3[tid] = d2;
    __syncthreads();
    if (tid == 0) {
        float a0 = 0.f, a1 = 0.f, a2 = 0.f, a3 = 0.f;
        for (int i = 0; i < 256; ++i) { a0 += r0[i]; a1 += r1[i]; a2 += r2[i]; a3 += r3[i]; }
        loss_out[0] = fminf(a0 / a1, a2 / a3);
    }
}

// ------------------------------------------------------------------- host ----

extern "C" void kernel_launch(void* const* d_in, const int* in_sizes, int n_in,
                              void* d_out, int out_size, void* d_ws, size_t ws_size,
                              hipStream_t stream) {
    const float* feat  = (const float*)d_in[0];
    const int*   tg1   = (const int*)  d_in[1];
    const int*   tg2   = (const int*)  d_in[2];
    const float* preW  = (const float*)d_in[3];
    const float* preb  = (const float*)d_in[4];
    const float* prelg = (const float*)d_in[5];
    const float* prelb = (const float*)d_in[6];
    const float* xlg   = (const float*)d_in[7];
    const float* xlb   = (const float*)d_in[8];
    const float* hlg   = (const float*)d_in[9];
    const float* hlb   = (const float*)d_in[10];
    const float* Wih   = (const float*)d_in[11];
    const float* Whh   = (const float*)d_in[12];
    const float* bih   = (const float*)d_in[13];
    const float* bhh   = (const float*)d_in[14];
    const float* projW = (const float*)d_in[15];
    const float* projb = (const float*)d_in[16];
    const float* ln1g  = (const float*)d_in[17];
    const float* ln1b  = (const float*)d_in[18];
    const float* tW1   = (const float*)d_in[19];
    const float* tb1   = (const float*)d_in[20];
    const float* tW2   = (const float*)d_in[21];
    const float* tb2   = (const float*)d_in[22];
    const float* ln2g  = (const float*)d_in[23];
    const float* ln2b  = (const float*)d_in[24];
    const float* chW1  = (const float*)d_in[25];
    const float* chb1  = (const float*)d_in[26];
    const float* chW2  = (const float*)d_in[27];
    const float* chb2  = (const float*)d_in[28];
    const float* flg   = (const float*)d_in[29];
    const float* flb   = (const float*)d_in[30];
    const float* headW = (const float*)d_in[31];
    const float* headb = (const float*)d_in[32];
    const float* cew   = (const float*)d_in[33];

    char* ws = (char*)d_ws;
    size_t off = 0;
    auto alloc = [&](size_t bytes) -> void* {
        void* p = ws + off;
        off = (off + bytes + 255) & ~(size_t)255;
        return p;
    };

    f16*   Wih16 = (f16*)  alloc((size_t)4 * 3072 * 1024 * 2);
    f16*   Whh16 = (f16*)  alloc((size_t)4 * 3072 * 1024 * 2);
    f16*   preWT = (f16*)  alloc((size_t)1024 * 2048 * 2);
    f16*   projT = (f16*)  alloc((size_t)512 * 1024 * 2);
    f16*   chW1T = (f16*)  alloc((size_t)2 * 2048 * 512 * 2);
    f16*   chW2T = (f16*)  alloc((size_t)2 * 512 * 2048 * 2);
    float* preact= (float*)alloc((size_t)512 * 16 * 1024 * 4);
    float* hbuf0 = (float*)alloc((size_t)4 * 16 * 1024 * 4);
    float* hbuf1 = (float*)alloc((size_t)4 * 16 * 1024 * 4);
    f16*   h16   = (f16*)  alloc((size_t)4 * 16 * 1024 * 2);
    float* tokens= (float*)alloc((size_t)16 * 4 * 512 * 4);
    f16*   t16   = (f16*)  alloc((size_t)64 * 512 * 2);
    f16*   u16   = (f16*)  alloc((size_t)64 * 2048 * 2);

    float* out      = (float*)d_out;
    float* out_loss = out + (size_t)16 * 512 * 22;

    // --- weight prep ---
    {
        long n = (long)4 * 3072 * 1024;
        cast_kernel<<<(unsigned)((n + 255) / 256), 256, 0, stream>>>(Wih, Wih16, n);
        cast_kernel<<<(unsigned)((n + 255) / 256), 256, 0, stream>>>(Whh, Whh16, n);
    }
    {
        long n = (long)2048 * 1024;
        transpose_cast_kernel<<<(unsigned)((n + 255) / 256), 256, 0, stream>>>(preW, preWT, 2048, 1024);
    }
    {
        long n = (long)1024 * 512;
        transpose_cast_kernel<<<(unsigned)((n + 255) / 256), 256, 0, stream>>>(projW, projT, 1024, 512);
    }
    for (int d = 0; d < 2; ++d) {
        long n = (long)512 * 2048;
        transpose_cast_kernel<<<(unsigned)((n + 255) / 256), 256, 0, stream>>>(
            chW1 + (size_t)d * 512 * 2048, chW1T + (size_t)d * 2048 * 512, 512, 2048);
        transpose_cast_kernel<<<(unsigned)((n + 255) / 256), 256, 0, stream>>>(
            chW2 + (size_t)d * 2048 * 512, chW2T + (size_t)d * 512 * 2048, 2048, 512);
    }
    zero_kernel<<<(4 * 16 * 1024 + 255) / 256, 256, 0, stream>>>(hbuf0, 4 * 16 * 1024);

    // --- pre-layer over all timesteps: TDM-staged WMMA GEMM + fused LN+GELU ---
    pre_kernel<<<512, 256, 0, stream>>>(feat, preWT, preb, prelg, prelb, preact);

    // --- recurrence ---
    for (int t = 0; t < 512; ++t) {
        float* hin  = (t & 1) ? hbuf1 : hbuf0;
        float* hout = (t & 1) ? hbuf0 : hbuf1;
        for (int i = 0; i < 4; ++i) {
            const float* x = (i == 0) ? (preact + (size_t)t * 16 * 1024)
                                      : (hout + (size_t)(i - 1) * 16 * 1024);
            gru_kernel<<<16, 128, 0, stream>>>(
                x, hin + (size_t)i * 16 * 1024, hout + (size_t)i * 16 * 1024,
                h16 + (size_t)i * 16 * 1024,
                Wih16 + (size_t)i * 3072 * 1024, Whh16 + (size_t)i * 3072 * 1024,
                bih + (size_t)i * 3072, bhh + (size_t)i * 3072,
                xlg + (size_t)i * 1024, xlb + (size_t)i * 1024,
                hlg + (size_t)i * 1024, hlb + (size_t)i * 1024);
        }
        proj_kernel<<<dim3(8, 4), 128, 0, stream>>>(h16, projT, projb, tokens);
        for (int d = 0; d < 2; ++d) {
            tokmix_kernel<<<16, 256, 0, stream>>>(tokens, ln1g + d * 512, ln1b + d * 512,
                                                  tW1 + d * 64, tb1 + d * 16,
                                                  tW2 + d * 64, tb2 + d * 4);
            lnrow_kernel<<<64, 32, 0, stream>>>(tokens, ln2g + d * 512, ln2b + d * 512, t16);
            mlp1_kernel<<<dim3(32, 4), 128, 0, stream>>>(
                t16, chW1T + (size_t)d * 2048 * 512, chb1 + (size_t)d * 2048, u16);
            mlp2_kernel<<<dim3(8, 4), 128, 0, stream>>>(
                u16, chW2T + (size_t)d * 512 * 2048, chb2 + (size_t)d * 512, tokens);
        }
        head_kernel<<<16, 256, 0, stream>>>(tokens, flg, flb, headW, headb, out, t);
    }

    loss_kernel<<<1, 256, 0, stream>>>(out, tg1, tg2, cew, out_loss);
}
